// MLPExtractor_56152402428133
// MI455X (gfx1250) — matrix operand order repaired and checked by reading, not compiled, not run on hardware
//
#include <hip/hip_runtime.h>
#include <hip/hip_bf16.h>
#include <math.h>

typedef __attribute__((ext_vector_type(16))) _Float16 v16h;
typedef __attribute__((ext_vector_type(8)))  float    v8f;

constexpr int kB    = 32;
constexpr int kN    = 90;
constexpr int kMAXN = 100;
constexpr int kH    = 64;
constexpr int kNP   = kN * kN;            // 8100 pairs
constexpr int kEFrow   = kH + kN;         // 154 floats per ef row
constexpr int kEFbatch = (kN + 1) * kEFrow;
constexpr int kTiles   = (kNP + 15) / 16; // 507 row-tiles of 16 pairs
constexpr float kNEG = -1e30f;

// Fast tanh: hardware v_tanh_f32 if available, else branch-free via v_exp_f32.
__device__ __forceinline__ float fast_tanh(float x) {
#if __has_builtin(__builtin_amdgcn_tanhf)
    return __builtin_amdgcn_tanhf(x);
#else
    // tanh(x) = 1 - 2/(exp(2x)+1); saturates correctly as exp -> 0 or inf.
    return 1.0f - 2.0f / (__expf(2.0f * x) + 1.0f);
#endif
}

// ---------------------------------------------------------------------------
// Kernel 1: per-batch precompute (sg, Apre, Bpre, counts, value MLP).
// ---------------------------------------------------------------------------
__global__ __launch_bounds__(256)
void prep_kernel(const float* __restrict__ ef,
                 const float* __restrict__ aW0, const float* __restrict__ ab0,
                 const float* __restrict__ cW0, const float* __restrict__ cb0,
                 const float* __restrict__ cW1, const float* __restrict__ cb1,
                 const float* __restrict__ cW2, const float* __restrict__ cb2,
                 const float* __restrict__ cW3, const float* __restrict__ cb3,
                 float* __restrict__ sg, float* __restrict__ Apre,
                 float* __restrict__ Bpre, float* __restrict__ counts,
                 float* __restrict__ valOut)
{
    const int b = blockIdx.x, tid = threadIdx.x;
    const float* efb = ef + (size_t)b * kEFbatch;

    __shared__ float xb[64], yb[64];
    __shared__ float red[256];

    if (tid < 64) xb[tid] = efb[tid];          // graph embedding
    __syncthreads();

    // sg = W0[:, :64] . g + b0
    if (tid < 64) {
        float s = ab0[tid];
        const float* wr = aW0 + tid * 192;
        #pragma unroll 8
        for (int k = 0; k < 64; ++k) s += wr[k] * xb[k];
        sg[b * 64 + tid] = s;
    }

    // Apre / Bpre: 90x64 each, K=64
    for (int idx = tid; idx < kN * 64; idx += 256) {
        int i = idx >> 6, c = idx & 63;
        const float* node = efb + (size_t)(1 + i) * kEFrow;
        const float* wa = aW0 + c * 192 + 64;
        const float* wb = aW0 + c * 192 + 128;
        float sa = 0.f, sb = 0.f;
        #pragma unroll 8
        for (int k = 0; k < 64; ++k) { float nv = node[k]; sa += wa[k] * nv; sb += wb[k] * nv; }
        Apre[((size_t)b * kN + i) * 64 + c] = sa;
        Bpre[((size_t)b * kN + i) * 64 + c] = sb;
    }

    // counts = sum(mask == 1)
    float cnt = 0.f;
    for (int p = tid; p < kNP; p += 256) {
        int i = p / kN, j = p - i * kN;
        cnt += (efb[(1 + i) * kEFrow + kH + j] == 1.0f) ? 1.f : 0.f;
    }
    red[tid] = cnt; __syncthreads();
    for (int s = 128; s > 0; s >>= 1) { if (tid < s) red[tid] += red[tid + s]; __syncthreads(); }
    if (tid == 0) counts[b] = red[0];

    // value MLP: 64 -> 64 -> 64 -> 64 -> 1 (tanh on first three)
    const float* Ws[3] = {cW0, cW1, cW2};
    const float* bs[3] = {cb0, cb1, cb2};
    for (int l = 0; l < 3; ++l) {
        if (tid < 64) {
            float s = bs[l][tid];
            const float* wr = Ws[l] + tid * 64;
            #pragma unroll 8
            for (int k = 0; k < 64; ++k) s += wr[k] * xb[k];
            yb[tid] = fast_tanh(s);
        }
        __syncthreads();
        if (tid < 64) xb[tid] = yb[tid];
        __syncthreads();
    }
    if (tid == 0) {
        float s = cb3[0];
        for (int k = 0; k < 64; ++k) s += cW3[k] * xb[k];
        valOut[b] = s;
    }
}

// ---------------------------------------------------------------------------
// Kernel 2: pair MLP core using v_wmma_f32_16x16x32_f16.
// 8 waves/block, one wave32 per 16-pair row-tile. Activation tile kept in
// LDS as f16 (A-fragment gathers become 2x ds_load_b128). Layouts per
// CDNA5 ISA §7.12.2 (C/D); A and B use a contiguous K order of this code's own.
// ---------------------------------------------------------------------------
__device__ __forceinline__ v8f wmma16(v16h a, v16h bm, v8f c) {
    return __builtin_amdgcn_wmma_f32_16x16x32_f16(
        /*neg_a=*/false, a, /*neg_b=*/false, bm,
        /*c_mod=*/(short)0, c, /*reuse_a=*/false, /*reuse_b=*/false);
}

__global__ __launch_bounds__(256)
void pair_kernel(const float* __restrict__ sg,
                 const float* __restrict__ Apre, const float* __restrict__ Bpre,
                 const float* __restrict__ aW1, const float* __restrict__ ab1,
                 const float* __restrict__ aW2, const float* __restrict__ ab2,
                 const float* __restrict__ aW3, const float* __restrict__ ab3,
                 float* __restrict__ gout)
{
    __shared__ _Float16 W1h[64 * 64];
    __shared__ _Float16 W2h[64 * 64];
    __shared__ float b1s[64], b2s[64], W3s[64];
    __shared__ _Float16 hbuf[8][16 * 64];   // one 16x64 f16 tile per wave

    const int tid = threadIdx.x;
    const int b = blockIdx.x;

    for (int idx = tid; idx < 4096; idx += 256) {
        W1h[idx] = (_Float16)aW1[idx];
        W2h[idx] = (_Float16)aW2[idx];
    }
    if (tid < 64) { b1s[tid] = ab1[tid]; b2s[tid] = ab2[tid]; W3s[tid] = aW3[tid]; }
    __syncthreads();

    const int w = tid >> 5, lane = tid & 31;
    const int tile = blockIdx.y * 8 + w;
    if (tile >= kTiles) return;            // wave-uniform; no barriers follow

    _Float16* hrow = hbuf[w];
    const float* sgb = sg + b * 64;
    const float* Ab  = Apre + (size_t)b * kN * 64;
    const float* Bb  = Bpre + (size_t)b * kN * 64;
    const float b3   = ab3[0];

    // h0[m][c] = tanh(sg[c] + Apre[i_m][c] + Bpre[j_m][c]); coalesced per row
    for (int t = 0; t < 32; ++t) {
        int m = t >> 1;
        int c = ((t & 1) << 5) + lane;
        int p = tile * 16 + m;
        if (p >= kNP) p = kNP - 1;         // clamp tail rows (results unused)
        int i = p / kN, j = p - i * kN;
        float v = sgb[c] + Ab[i * 64 + c] + Bb[j * 64 + c];
        hrow[m * 64 + c] = (_Float16)fast_tanh(v);
    }

    // Layers 1 and 2: Y(16x64) = X(16x64) * W^T, K=64 -> 2 k-chunks of 32
    #pragma unroll
    for (int layer = 0; layer < 2; ++layer) {
        const _Float16* W = layer ? W2h : W1h;
        const float* bias = layer ? b2s : b1s;

        const int m    = lane & 15;
        const int half = lane >> 4;

        // A fragments (lane%16=row, K contiguous (this code's order, same as B): lanes 0-15 K=0..15, lanes 16-31 K=16..31)
        v16h a0, a1;
        #pragma unroll
        for (int t = 0; t < 16; ++t) {
            int cofs = 16 * half + t;
            a0[t] = hrow[m * 64 + cofs];
            a1[t] = hrow[m * 64 + 32 + cofs];
        }

        const int n  = lane & 15;          // output column within n-tile
        const int kh = lane >> 4;          // K half for B layout
        #pragma unroll
        for (int nt = 0; nt < 4; ++nt) {
            const int o = nt * 16 + n;     // output channel
            // B fragments (K-major: lanes 0-15 K=0..15, lanes 16-31 K=16..31)
            v16h bf0, bf1;
            #pragma unroll
            for (int t = 0; t < 16; ++t) {
                bf0[t] = W[o * 64 + kh * 16 + t];
                bf1[t] = W[o * 64 + 32 + kh * 16 + t];
            }
            v8f acc = {};
            acc = wmma16(a0, bf0, acc);
            acc = wmma16(a1, bf1, acc);
            // C/D layout: element v -> row v + 8*(lane/16), col = o
            float bo = bias[o];
            #pragma unroll
            for (int v = 0; v < 8; ++v) {
                int mm = v + 8 * half;
                hrow[mm * 64 + o] = (_Float16)fast_tanh(acc[v] + bo);
            }
        }
    }

    // Layer 3: logit[m] = b3 + sum_c W3[c]*h2[m][c]; 2 lanes per row
    {
        const int m3 = lane & 15, ch = lane >> 4;
        float part = 0.f;
        #pragma unroll 8
        for (int c = 0; c < 32; ++c)
            part += W3s[ch * 32 + c] * (float)hrow[m3 * 64 + ch * 32 + c];
        part += __shfl_xor(part, 16, 32);
        int p = tile * 16 + m3;
        if (lane < 16 && p < kNP)
            gout[(size_t)b * (kMAXN * kMAXN) + p] = part + b3;  // stage logits
    }
}

// ---------------------------------------------------------------------------
// Kernel 3: masked softmax with phantom-slot correction + padded output write.
// ---------------------------------------------------------------------------
__global__ __launch_bounds__(256)
void softmax_kernel(const float* __restrict__ ef, const float* __restrict__ counts,
                    float* __restrict__ gout)
{
    __shared__ float l[kNP];
    __shared__ unsigned char mb[kNP];
    __shared__ float red[256];

    const int b = blockIdx.x, tid = threadIdx.x;
    const float* efb = ef + (size_t)b * kEFbatch;
    float* go = gout + (size_t)b * (kMAXN * kMAXN);

    float dim = 0.f;
    for (int q = 0; q < kB; ++q) dim = fmaxf(dim, counts[q]);
    const float cnt = counts[b];

    for (int p = tid; p < kNP; p += 256) {
        l[p] = go[p];
        int i = p / kN, j = p - i * kN;
        mb[p] = (efb[(1 + i) * kEFrow + kH + j] == 1.0f) ? 1 : 0;
    }
    __syncthreads();

    // masked max
    float lm = kNEG;
    for (int p = tid; p < kNP; p += 256) if (mb[p]) lm = fmaxf(lm, l[p]);
    red[tid] = lm; __syncthreads();
    for (int s = 128; s > 0; s >>= 1) { if (tid < s) red[tid] = fmaxf(red[tid], red[tid + s]); __syncthreads(); }
    float maxv = red[0]; __syncthreads();
    const float m = (cnt < dim) ? fmaxf(maxv, 0.f) : maxv;

    // masked exp-sum (hardware v_exp_f32)
    float s = 0.f;
    for (int p = tid; p < kNP; p += 256) if (mb[p]) s += __expf(l[p] - m);
    red[tid] = s; __syncthreads();
    for (int st = 128; st > 0; st >>= 1) { if (tid < st) red[tid] += red[tid + st]; __syncthreads(); }
    const float denom = red[0] + (dim - cnt) * __expf(-m);
    __syncthreads();

    // write 100x100 padded pi (also zeroes padding / overwrites staged logits)
    for (int q = tid; q < kMAXN * kMAXN; q += 256) {
        int r = q / kMAXN, c = q - r * kMAXN;
        float o = 0.f;
        if (r < kN && c < kN) {
            int p = r * kN + c;
            if (mb[p]) o = __expf(l[p] - m) / denom;
        }
        go[q] = o;
    }
}

// ---------------------------------------------------------------------------
extern "C" void kernel_launch(void* const* d_in, const int* in_sizes, int n_in,
                              void* d_out, int out_size, void* d_ws, size_t ws_size,
                              hipStream_t stream)
{
    const float* ef  = (const float*)d_in[0];
    const float* aW0 = (const float*)d_in[1];
    const float* ab0 = (const float*)d_in[2];
    const float* aW1 = (const float*)d_in[3];
    const float* ab1 = (const float*)d_in[4];
    const float* aW2 = (const float*)d_in[5];
    const float* ab2 = (const float*)d_in[6];
    const float* aW3 = (const float*)d_in[7];
    const float* ab3 = (const float*)d_in[8];
    const float* cW0 = (const float*)d_in[9];
    const float* cb0 = (const float*)d_in[10];
    const float* cW1 = (const float*)d_in[11];
    const float* cb1 = (const float*)d_in[12];
    const float* cW2 = (const float*)d_in[13];
    const float* cb2 = (const float*)d_in[14];
    const float* cW3 = (const float*)d_in[15];
    const float* cb3 = (const float*)d_in[16];

    float* out = (float*)d_out;

    // workspace layout (floats): sg | Apre | Bpre | counts   (~1.5 MB)
    float* ws     = (float*)d_ws;
    float* sg     = ws;
    float* Apre   = sg + kB * 64;
    float* Bpre   = Apre + (size_t)kB * kN * 64;
    float* counts = Bpre + (size_t)kB * kN * 64;

    float* valOut = out + (size_t)kB * kMAXN * kMAXN;  // value tail of d_out

    prep_kernel<<<dim3(kB), dim3(256), 0, stream>>>(
        ef, aW0, ab0, cW0, cb0, cW1, cb1, cW2, cb2, cW3, cb3,
        sg, Apre, Bpre, counts, valOut);

    pair_kernel<<<dim3(kB, (kTiles + 7) / 8), dim3(256), 0, stream>>>(
        sg, Apre, Bpre, aW1, ab1, aW2, ab2, aW3, ab3, out);

    softmax_kernel<<<dim3(kB), dim3(256), 0, stream>>>(ef, counts, out);
}
